// EdgeUpdateLayer_6846177870418
// MI455X (gfx1250) — matrix operand order repaired and checked by reading, not compiled, not run on hardware
//
#include <hip/hip_runtime.h>
#include <hip/hip_bf16.h>

typedef __attribute__((ext_vector_type(16))) _Float16 v16h;
typedef __attribute__((ext_vector_type(8)))  float    v8f;
typedef __attribute__((ext_vector_type(2)))  _Float16 h2;

#define IN_DIM   192
#define D_DIM    64
#define TILE_M   128
#define XSTRIDE  200   // 192 halves + 8 pad (row = 400B, 4B aligned)

// ----- ws layout (bytes) -----
// [0,      73728)  W1T f16 [192][192]  W1T[k*192+n] = W1[n*192+k]
// [73728,  98304)  W2T f16 [192][64]   W2T[k*64 +n] = W2[n*192+k]
// [98304,  99072)  gsum   f32[192]
// [99072,  99840)  gsumsq f32[192]
// [99840, 100608)  bn_a   f32[192]  (= gamma * rstd)
// [100608,101376)  bn_c   f32[192]  (= beta - mu * gamma * rstd)

__device__ __forceinline__ v8f wmma16(v16h a, v16h b, v8f c) {
    return __builtin_amdgcn_wmma_f32_16x16x32_f16(false, a, false, b, (short)0, c, false, false);
}

union AFrag { v16h v; h2 p[8]; };

// 16-bit A fragment 16x32 (MxK): lane L holds row m = mbase + (L&15);
// half pair r: k = kbase + 2r + (r>=4 ? 8 : 0) + 8*(L>>4)
__device__ __forceinline__ v16h load_a_frag(const _Float16* xs, int mbase, int lane, int kbase) {
    const _Float16* rp = xs + (mbase + (lane & 15)) * XSTRIDE + kbase + ((lane >> 4) << 3);
    AFrag f;
    f.p[0] = *(const h2*)(rp + 0);
    f.p[1] = *(const h2*)(rp + 2);
    f.p[2] = *(const h2*)(rp + 4);
    f.p[3] = *(const h2*)(rp + 6);
    f.p[4] = *(const h2*)(rp + 16);
    f.p[5] = *(const h2*)(rp + 18);
    f.p[6] = *(const h2*)(rp + 20);
    f.p[7] = *(const h2*)(rp + 22);
    return f.v;
}

// 16-bit B fragment 32x16 (KxN): lane L holds column n = nbase + (L&15);
// element s: k = kbase + 8*(L>>4) + s (s<8), kbase + 16 + 8*(L>>4) + (s-8) (s>=8).
__device__ __forceinline__ v16h load_b_frag(const _Float16* wt, int ldw, int kbase, int nbase, int lane) {
    const _Float16* cp = wt + (size_t)(kbase + ((lane >> 4) << 3)) * ldw + nbase + (lane & 15);
    v16h b;
    #pragma unroll
    for (int s = 0; s < 16; ++s)
        b[s] = cp[(size_t)((s < 8) ? s : (s + 8)) * ldw];
    return b;
}

// Stage x = [ni+nj | abs(ni-nj) | edge_feats] as f16 into LDS.
// 256 threads, 2 threads per row, 32 feats per thread. Rows past E zero-padded.
__device__ __forceinline__ void stage_x(const float* __restrict__ nodef,
                                        const float* __restrict__ edgef,
                                        const long long* __restrict__ ei,
                                        long long E, long long ebase,
                                        _Float16* xs, int tid) {
    int row = tid >> 1;
    int seg = tid & 1;
    long long e = ebase + row;
    _Float16* r = xs + row * XSTRIDE;
    int d0 = seg * 32;
    if (e < E) {
        long long i = ei[e];
        long long j = ei[E + e];
        const float* ni = nodef + (size_t)i * D_DIM;
        const float* nj = nodef + (size_t)j * D_DIM;
        const float* ef = edgef + (size_t)e * D_DIM;
        #pragma unroll
        for (int d = 0; d < 32; d += 4) {
            float4 a = *(const float4*)(ni + d0 + d);
            float4 b = *(const float4*)(nj + d0 + d);
            float4 c = *(const float4*)(ef + d0 + d);
            h2 t;
            t.x = (_Float16)(a.x + b.x); t.y = (_Float16)(a.y + b.y);
            *(h2*)(r + d0 + d) = t;
            t.x = (_Float16)(a.z + b.z); t.y = (_Float16)(a.w + b.w);
            *(h2*)(r + d0 + d + 2) = t;
            t.x = (_Float16)fabsf(a.x - b.x); t.y = (_Float16)fabsf(a.y - b.y);
            *(h2*)(r + 64 + d0 + d) = t;
            t.x = (_Float16)fabsf(a.z - b.z); t.y = (_Float16)fabsf(a.w - b.w);
            *(h2*)(r + 64 + d0 + d + 2) = t;
            t.x = (_Float16)c.x; t.y = (_Float16)c.y;
            *(h2*)(r + 128 + d0 + d) = t;
            t.x = (_Float16)c.z; t.y = (_Float16)c.w;
            *(h2*)(r + 128 + d0 + d + 2) = t;
        }
    } else {
        h2 z; z.x = (_Float16)0.f; z.y = (_Float16)0.f;
        #pragma unroll
        for (int d = 0; d < 32; d += 2) {
            *(h2*)(r + d0 + d)       = z;
            *(h2*)(r + 64 + d0 + d)  = z;
            *(h2*)(r + 128 + d0 + d) = z;
        }
    }
}

// ---------------- kernel 0: weight transpose/convert + zero stats ----------------
__global__ __launch_bounds__(256) void prep_kernel(const float* __restrict__ W1,
                                                   const float* __restrict__ W2,
                                                   _Float16* __restrict__ w1t,
                                                   _Float16* __restrict__ w2t,
                                                   float* __restrict__ gsum,
                                                   float* __restrict__ gsq) {
    int idx = blockIdx.x * blockDim.x + threadIdx.x;
    const int N1 = IN_DIM * IN_DIM;           // 36864
    const int N2 = IN_DIM * D_DIM;            // 12288
    if (idx < N1) {
        int k = idx / IN_DIM, n = idx % IN_DIM;
        w1t[idx] = (_Float16)W1[n * IN_DIM + k];
    } else if (idx < N1 + N2) {
        int t = idx - N1;
        int k = t / D_DIM, n = t % D_DIM;
        w2t[t] = (_Float16)W2[n * IN_DIM + k];
    } else if (idx < N1 + N2 + 2 * IN_DIM) {
        int t = idx - (N1 + N2);
        if (t < IN_DIM) gsum[t] = 0.f; else gsq[t - IN_DIM] = 0.f;
    }
}

// ---------------- kernel 1: GEMM1 + column sum / sumsq of h ----------------
__global__ __launch_bounds__(256) void stats_kernel(const float* __restrict__ nodef,
                                                    const float* __restrict__ edgef,
                                                    const long long* __restrict__ ei,
                                                    const float* __restrict__ b1,
                                                    const _Float16* __restrict__ w1t,
                                                    float* __restrict__ gsum,
                                                    float* __restrict__ gsq,
                                                    long long E) {
    __shared__ _Float16 xs[TILE_M * XSTRIDE];
    __shared__ float ssum[IN_DIM];
    __shared__ float ssq[IN_DIM];
    int tid = threadIdx.x;
    if (tid < IN_DIM) { ssum[tid] = 0.f; ssq[tid] = 0.f; }
    long long ebase = (long long)blockIdx.x * TILE_M;
    stage_x(nodef, edgef, ei, E, ebase, xs, tid);
    __syncthreads();

    int lane = tid & 31;
    int wave = tid >> 5;
    int mbase = wave * 16;
    int ncol = lane & 15;
    int rbase = (lane >> 4) * 8;                 // rows: m = mbase + rbase + jj

    // Scalar-uniform tail detection: all blocks except (at most) the last are full.
    int rem = (int)((E - ebase < (long long)TILE_M) ? (E - ebase) : (long long)TILE_M);
    bool fullTile = (rem == TILE_M);

    v16h afr[6];
    #pragma unroll
    for (int kt = 0; kt < 6; ++kt) afr[kt] = load_a_frag(xs, mbase, lane, kt * 32);

    #pragma unroll
    for (int nt = 0; nt < 12; ++nt) {
        v8f acc = {};
        #pragma unroll
        for (int kt = 0; kt < 6; ++kt)
            acc = wmma16(afr[kt], load_b_frag(w1t, IN_DIM, kt * 32, nt * 16, lane), acc);
        float bb = b1[nt * 16 + ncol];
        float s1 = 0.f, s2 = 0.f;
        if (fullTile) {                          // s_cbranch: no per-element guards
            #pragma unroll
            for (int jj = 0; jj < 8; ++jj) {
                float v = acc[jj] + bb;
                s1 += v;
                s2 += v * v;
            }
        } else {
            int myrem = rem - mbase - rbase;     // valid jj: jj < myrem (32-bit)
            #pragma unroll
            for (int jj = 0; jj < 8; ++jj) {
                if (jj < myrem) {
                    float v = acc[jj] + bb;
                    s1 += v;
                    s2 += v * v;
                }
            }
        }
        atomicAdd(&ssum[nt * 16 + ncol], s1);
        atomicAdd(&ssq[nt * 16 + ncol], s2);
    }
    __syncthreads();
    if (tid < IN_DIM) {
        atomicAdd(&gsum[tid], ssum[tid]);
        atomicAdd(&gsq[tid], ssq[tid]);
    }
}

// ---------------- kernel 2: finalize BN scale/shift ----------------
__global__ void finalize_kernel(const float* __restrict__ gamma,
                                const float* __restrict__ beta,
                                const float* __restrict__ gsum,
                                const float* __restrict__ gsq,
                                float* __restrict__ bn_a,
                                float* __restrict__ bn_c,
                                float invE) {
    int t = threadIdx.x;
    if (t < IN_DIM) {
        float mu  = gsum[t] * invE;
        float var = gsq[t] * invE - mu * mu;
        float rstd = rsqrtf(var + 1e-5f);
        float a = gamma[t] * rstd;
        bn_a[t] = a;
        bn_c[t] = beta[t] - mu * a;
    }
}

// ---------------- kernel 3: GEMM1 -> BN -> ReLU -> GEMM2 -> out ----------------
__global__ __launch_bounds__(256) void main_kernel(const float* __restrict__ nodef,
                                                   const float* __restrict__ edgef,
                                                   const long long* __restrict__ ei,
                                                   const float* __restrict__ b1,
                                                   const float* __restrict__ b2,
                                                   const _Float16* __restrict__ w1t,
                                                   const _Float16* __restrict__ w2t,
                                                   const float* __restrict__ bn_a,
                                                   const float* __restrict__ bn_c,
                                                   float* __restrict__ out,
                                                   long long E) {
    __shared__ _Float16 xs[TILE_M * XSTRIDE];
    int tid = threadIdx.x;
    long long ebase = (long long)blockIdx.x * TILE_M;
    stage_x(nodef, edgef, ei, E, ebase, xs, tid);
    __syncthreads();

    int lane = tid & 31;
    int wave = tid >> 5;
    int mbase = wave * 16;
    int ncol = lane & 15;
    int rbase = (lane >> 4) * 8;

    int rem = (int)((E - ebase < (long long)TILE_M) ? (E - ebase) : (long long)TILE_M);
    bool fullTile = (rem == TILE_M);

    // GEMM1: keep all 12 N-tiles live (k-outer) so A frags are loaded once per kt.
    v8f acc[12] = {};
    #pragma unroll
    for (int kt = 0; kt < 6; ++kt) {
        v16h a = load_a_frag(xs, mbase, lane, kt * 32);
        #pragma unroll
        for (int nt = 0; nt < 12; ++nt)
            acc[nt] = wmma16(a, load_b_frag(w1t, IN_DIM, kt * 32, nt * 16, lane), acc[nt]);
    }

    // BN + ReLU epilogue, convert to f16, overwrite this wave's own 16 LDS rows.
    // (Each wave only ever touches rows [mbase, mbase+16) after the barrier.)
    #pragma unroll
    for (int nt = 0; nt < 12; ++nt) {
        int n = nt * 16 + ncol;
        float bb = b1[n];
        float aa = bn_a[n];
        float cc = bn_c[n];
        #pragma unroll
        for (int jj = 0; jj < 8; ++jj) {
            float hv = acc[nt][jj] + bb;
            float y  = hv * aa + cc;
            y = fmaxf(y, 0.f);
            xs[(mbase + rbase + jj) * XSTRIDE + n] = (_Float16)y;
        }
    }

    // GEMM2: y[16x192] @ W2^T[192x64]
    v8f acc2[4] = {};
    #pragma unroll
    for (int kt = 0; kt < 6; ++kt) {
        v16h a = load_a_frag(xs, mbase, lane, kt * 32);
        #pragma unroll
        for (int nt = 0; nt < 4; ++nt)
            acc2[nt] = wmma16(a, load_b_frag(w2t, D_DIM, kt * 32, nt * 16, lane), acc2[nt]);
    }

    // Output: base pointer + immediate row offsets on the (uniform) fast path.
    float* op = out + (size_t)(ebase + mbase + rbase) * D_DIM + ncol;
    if (fullTile) {
        #pragma unroll
        for (int nt = 0; nt < 4; ++nt) {
            float bb = b2[nt * 16 + ncol];
            #pragma unroll
            for (int jj = 0; jj < 8; ++jj)
                op[(size_t)jj * D_DIM + nt * 16] = acc2[nt][jj] + bb;
        }
    } else {
        int myrem = rem - mbase - rbase;          // valid jj: jj < myrem (32-bit)
        #pragma unroll
        for (int nt = 0; nt < 4; ++nt) {
            float bb = b2[nt * 16 + ncol];
            #pragma unroll
            for (int jj = 0; jj < 8; ++jj)
                if (jj < myrem)
                    op[(size_t)jj * D_DIM + nt * 16] = acc2[nt][jj] + bb;
        }
    }
}

extern "C" void kernel_launch(void* const* d_in, const int* in_sizes, int n_in,
                              void* d_out, int out_size, void* d_ws, size_t ws_size,
                              hipStream_t stream) {
    const float*     nodef = (const float*)d_in[0];
    const float*     edgef = (const float*)d_in[1];
    const long long* ei    = (const long long*)d_in[2];   // int64 [2,E]
    const float*     W1    = (const float*)d_in[3];
    const float*     b1    = (const float*)d_in[4];
    const float*     gamma = (const float*)d_in[5];
    const float*     beta  = (const float*)d_in[6];
    const float*     W2    = (const float*)d_in[7];
    const float*     b2    = (const float*)d_in[8];
    float*           out   = (float*)d_out;

    long long E = (long long)in_sizes[2] / 2;

    char* ws = (char*)d_ws;
    _Float16* w1t  = (_Float16*)(ws);
    _Float16* w2t  = (_Float16*)(ws + 73728);
    float*    gsum = (float*)(ws + 98304);
    float*    gsq  = (float*)(ws + 99072);
    float*    bn_a = (float*)(ws + 99840);
    float*    bn_c = (float*)(ws + 100608);

    const int prep_total = IN_DIM * IN_DIM + IN_DIM * D_DIM + 2 * IN_DIM;
    prep_kernel<<<(prep_total + 255) / 256, 256, 0, stream>>>(W1, W2, w1t, w2t, gsum, gsq);

    unsigned nblocks = (unsigned)((E + TILE_M - 1) / TILE_M);
    stats_kernel<<<nblocks, 256, 0, stream>>>(nodef, edgef, ei, b1, w1t, gsum, gsq, E);
    finalize_kernel<<<1, IN_DIM, 0, stream>>>(gamma, beta, gsum, gsq, bn_a, bn_c, 1.0f / (float)E);
    main_kernel<<<nblocks, 256, 0, stream>>>(nodef, edgef, ei, b1, b2, w1t, w2t, bn_a, bn_c, out, E);
}
